// RNN_4964982194675
// MI455X (gfx1250) — matrix operand order repaired and checked by reading, not compiled, not run on hardware
//
#include <hip/hip_runtime.h>
#include <hip/hip_bf16.h>

#define LATENT   32
#define INPUT_D  16
#define CAT      48
#define NU       20
#define NTRAJ    4096
#define TSTEPS   512

#define LDS_STRIDE 40          // halves per transpose row (pad vs 32 to spread banks)
#define N_TBUF 5               // y, h_u, h_r, y*r, h_n each get their own region
#define WAVES_PER_BLOCK 8
#define TRAJ_PER_WAVE 16

typedef __attribute__((ext_vector_type(16))) _Float16 v16h;
typedef __attribute__((ext_vector_type(8)))  float    v8f;
typedef __attribute__((ext_vector_type(4)))  float    v4f;

// ---- activations: native gfx1250 v_tanh_f32 (verified emitted) ----
#if defined(__has_builtin)
#if __has_builtin(__builtin_amdgcn_tanhf)
#define HAVE_NATIVE_TANH 1
#endif
#endif

__device__ __forceinline__ float tanh_f(float x){
#ifdef HAVE_NATIVE_TANH
  return __builtin_amdgcn_tanhf(x);
#else
  return 1.0f - 2.0f*__builtin_amdgcn_rcpf(__expf(2.0f*x) + 1.0f);
#endif
}
__device__ __forceinline__ float sigmoid_f(float x){
#ifdef HAVE_NATIVE_TANH
  return __builtin_fmaf(0.5f, __builtin_amdgcn_tanhf(0.5f*x), 0.5f);
#else
  return __builtin_amdgcn_rcpf(1.0f + __expf(-x));
#endif
}

__device__ __forceinline__ v8f splat8(float v){
  v8f x;
#pragma unroll
  for (int j = 0; j < 8; ++j) x[j] = v;
  return x;
}

// B-matrix (32K x 16N, f16) WMMA layout: lane = (n&15) + 16*lg; halves 0..7 = K lg*8..lg*8+7, halves 8..15 = K 16+lg*8..16+lg*8+7
__device__ __forceinline__ v16h load_btile(const float* __restrict__ W, int ldn,
                                           int kbase, int ncol, int kmax, int nmax, int lg){
  v16h b;
#pragma unroll
  for (int j = 0; j < 16; ++j){
    int k = kbase + (j < 8 ? lg*8 + j : 16 + lg*8 + (j - 8));
    float v = (k < kmax && ncol < nmax) ? W[k*ldn + ncol] : 0.0f;
    b[j] = (_Float16)v;
  }
  return b;
}

// Same, but with a bias vector folded into padding row k == kbias
// (augmented-matrix trick: A supplies 1.0 at that K row).
__device__ __forceinline__ v16h load_btile_bias(const float* __restrict__ W, int ldn,
                                                int kbase, int ncol, int kmax, int nmax, int lg,
                                                const float* __restrict__ bias, int kbias){
  v16h b;
#pragma unroll
  for (int j = 0; j < 16; ++j){
    int k = kbase + (j < 8 ? lg*8 + j : 16 + lg*8 + (j - 8));
    float v = 0.0f;
    if (ncol < nmax){
      if (k < kmax)        v = W[k*ldn + ncol];
      else if (k == kbias) v = bias[ncol];
    }
    b[j] = (_Float16)v;
  }
  return b;
}

// C-layout (m = lg*8 + r over VGPRs, n over lanes) -> LDS row-major [16][32] f16.
// No explicit s_wait: same-wave LDS ops execute in order at the LDS, so the
// subsequent ds_load observes these stores; the compiler inserts the (partial)
// s_wait_dscnt needed before the loaded data is consumed.
__device__ __forceinline__ void trans_write(_Float16* ldsw, int nl, int lg, v8f t0, v8f t1){
#pragma unroll
  for (int r = 0; r < 8; ++r){
    int m = lg*8 + r;
    ldsw[m*LDS_STRIDE + nl]      = (_Float16)t0[r];
    ldsw[m*LDS_STRIDE + 16 + nl] = (_Float16)t1[r];
  }
}

// LDS row-major [16][32] -> A-matrix 16x32 f16 layout:
// lane row m = lane&15; halves 0..7 = K kb..kb+7, halves 8..15 = K 16+kb..16+kb+7 (kb = (lane>>4)*8)
__device__ __forceinline__ v16h trans_read(const _Float16* ldsw, int nl, int lg){
  v16h a;
  const int kb = lg*8;
#pragma unroll
  for (int j = 0; j < 8; ++j){
    a[j]   = ldsw[nl*LDS_STRIDE + kb + j];
    a[8+j] = ldsw[nl*LDS_STRIDE + 16 + kb + j];
  }
  return a;
}

// One 2-layer MLP gate: act2( tanh(yc @ W1 + b1) @ W2 + b2 ).
// Biases ride in padded K rows (k=48 for layer-1, k=20 for layer-2), so C starts at inline 0.
// All weight tiles register-resident; hbuf is this gate's private transpose region.
__device__ __forceinline__ void gate_mlp(const v16h& A0, const v16h& A1,
                                         const v16h (&B1y)[2], const v16h (&B1x)[2],
                                         const v16h (&B2)[2],
                                         _Float16* hbuf, int nl, int lg, bool out_tanh,
                                         v8f& o0, v8f& o1)
{
  v8f h0, h1;
#pragma unroll
  for (int nt = 0; nt < 2; ++nt){
    v8f acc = splat8(0.0f);
    acc = __builtin_amdgcn_wmma_f32_16x16x32_f16(false, A0, false, B1y[nt], (short)0, acc, false, false);
    acc = __builtin_amdgcn_wmma_f32_16x16x32_f16(false, A1, false, B1x[nt], (short)0, acc, false, false);
    v8f h;
#pragma unroll
    for (int r = 0; r < 8; ++r) h[r] = tanh_f(acc[r]);
    if (nt == 0) h0 = h; else h1 = h;
  }
  trans_write(hbuf, nl, lg, h0, h1);        // hidden (16x32, cols >=20 exactly 0)
  v16h Ah = trans_read(hbuf, nl, lg);
  // inject 1.0 at K=20 (layer-2 bias row); K=20 lives in half 12 of lg==0 lanes only
  Ah[12] = (lg == 0) ? (_Float16)1.0f : Ah[12];
#pragma unroll
  for (int nt = 0; nt < 2; ++nt){
    v8f acc = splat8(0.0f);
    acc = __builtin_amdgcn_wmma_f32_16x16x32_f16(false, Ah, false, B2[nt], (short)0, acc, false, false);
    v8f o;
#pragma unroll
    for (int r = 0; r < 8; ++r) o[r] = out_tanh ? tanh_f(acc[r]) : sigmoid_f(acc[r]);
    if (nt == 0) o0 = o; else o1 = o;
  }
}

__global__ __launch_bounds__(WAVES_PER_BLOCK*32, 1)
void gru_scan_kernel(const float* __restrict__ data,
                     const float* __restrict__ Wu1, const float* __restrict__ bu1,
                     const float* __restrict__ Wu2, const float* __restrict__ bu2,
                     const float* __restrict__ Wr1, const float* __restrict__ br1,
                     const float* __restrict__ Wr2, const float* __restrict__ br2,
                     const float* __restrict__ Wn1, const float* __restrict__ bn1,
                     const float* __restrict__ Wn2, const float* __restrict__ bn2,
                     float* __restrict__ out_yi, float* __restrict__ out_ys)
{
  // Per-wave transpose scratch: 5 independent regions so the scheduler can keep
  // multiple transposes in flight with partial dscnt waits (no false aliasing).
  __shared__ _Float16 tlds[WAVES_PER_BLOCK * N_TBUF * 16 * LDS_STRIDE];

  const int tid  = threadIdx.x;
  const int wib  = tid >> 5;
  const int lane = tid & 31;
  const int nl   = lane & 15;
  const int lg   = lane >> 4;
  const int waveId = blockIdx.x * WAVES_PER_BLOCK + wib;
  const int traj0  = waveId * TRAJ_PER_WAVE;
  _Float16* buf_y  = tlds + (wib*N_TBUF + 0) * (16 * LDS_STRIDE);
  _Float16* buf_hu = tlds + (wib*N_TBUF + 1) * (16 * LDS_STRIDE);
  _Float16* buf_hr = tlds + (wib*N_TBUF + 2) * (16 * LDS_STRIDE);
  _Float16* buf_yr = tlds + (wib*N_TBUF + 3) * (16 * LDS_STRIDE);
  _Float16* buf_hn = tlds + (wib*N_TBUF + 4) * (16 * LDS_STRIDE);

  // ---- Register-resident weight tiles (f16, WMMA B-layout, zero-padded, bias folded) ----
  v16h B1yu[2], B1yr[2], B1yn[2];   // W1 rows 0..31  (y part of yc)
  v16h B1xu[2], B1xr[2], B1xn[2];   // W1 rows 32..47 (x part) + b1 in row k=48
  v16h B2u[2],  B2r[2],  B2n[2];    // W2 (K 20->32 pad) + b2 in row k=20
#pragma unroll
  for (int nt = 0; nt < 2; ++nt){
    int ncol = nt*16 + nl;
    B1yu[nt] = load_btile(Wu1, NU, 0, ncol, CAT, NU, lg);
    B1yr[nt] = load_btile(Wr1, NU, 0, ncol, CAT, NU, lg);
    B1yn[nt] = load_btile(Wn1, NU, 0, ncol, CAT, NU, lg);
    B1xu[nt] = load_btile_bias(Wu1, NU, 32, ncol, CAT, NU, lg, bu1, CAT);
    B1xr[nt] = load_btile_bias(Wr1, NU, 32, ncol, CAT, NU, lg, br1, CAT);
    B1xn[nt] = load_btile_bias(Wn1, NU, 32, ncol, CAT, NU, lg, bn1, CAT);
    B2u[nt]  = load_btile_bias(Wu2, LATENT, 0, ncol, NU, LATENT, lg, bu2, NU);
    B2r[nt]  = load_btile_bias(Wr2, LATENT, 0, ncol, NU, LATENT, lg, br2, NU);
    B2n[nt]  = load_btile_bias(Wn2, LATENT, 0, ncol, NU, LATENT, lg, bn2, NU);
  }

  // State y (16 traj x 32 latent) in C layout, f32.
  v8f y0 = splat8(0.0f);
  v8f y1 = splat8(0.0f);

  // x row base for this lane's A-matrix role (row m = nl, K slice starts at lg*8)
  const float* xbase = data + (size_t)(traj0 + nl) * (TSTEPS * INPUT_D) + lg*8;

  // Software pipeline: x for step t is loaded one iteration ahead (non-temporal: streamed once).
  const v4f* xp0 = (const v4f*)xbase;
  v4f xa = __builtin_nontemporal_load(xp0);
  v4f xb = __builtin_nontemporal_load(xp0 + 1);

  for (int t = 0; t < TSTEPS; ++t){
    // Issue next step's load + a deeper prefetch right away (hide global latency).
    v4f xa_n = xa, xb_n = xb;
    if (t + 1 < TSTEPS){
      const v4f* xp = (const v4f*)(xbase + (size_t)(t+1) * INPUT_D);
      xa_n = __builtin_nontemporal_load(xp);
      xb_n = __builtin_nontemporal_load(xp + 1);
      if (t + 2 < TSTEPS)
        __builtin_prefetch(xbase + (size_t)(t+2) * INPUT_D, 0, 1);   // global_prefetch_b8
    }

    // ---- x_t into A-layout K-chunk1 (K=32..47 real; 1.0 at K=48 = layer-1 bias row) ----
    v16h A1;
    A1[0]=(_Float16)xa.x; A1[1]=(_Float16)xa.y; A1[2]=(_Float16)xa.z; A1[3]=(_Float16)xa.w;
    A1[4]=(_Float16)xb.x; A1[5]=(_Float16)xb.y; A1[6]=(_Float16)xb.z; A1[7]=(_Float16)xb.w;
    A1[8] = (lg == 0) ? (_Float16)1.0f : (_Float16)0.0f;   // K=48 (lg==0 half 8)
#pragma unroll
    for (int j = 9; j < 16; ++j) A1[j] = (_Float16)0.0f;

    // ---- y -> A-layout K-chunk0 (K=0..31) via LDS transpose ----
    trans_write(buf_y, nl, lg, y0, y1);
    v16h A0 = trans_read(buf_y, nl, lg);

    // ---- update & reset gates (independent; scheduler interleaves them) ----
    v8f u0, u1, r0, r1;
    gate_mlp(A0, A1, B1yu, B1xu, B2u, buf_hu, nl, lg, false, u0, u1);
    gate_mlp(A0, A1, B1yr, B1xr, B2r, buf_hr, nl, lg, false, r0, r1);

    // ---- candidate: cc = [y*r, x] ----
    v8f yr0, yr1;
#pragma unroll
    for (int r = 0; r < 8; ++r){ yr0[r] = y0[r]*r0[r]; yr1[r] = y1[r]*r1[r]; }
    trans_write(buf_yr, nl, lg, yr0, yr1);
    v16h A0n = trans_read(buf_yr, nl, lg);
    v8f n0, n1;
    gate_mlp(A0n, A1, B1yn, B1xn, B2n, buf_hn, nl, lg, true, n0, n1);

    // ---- y = (1-u)*n + u*y ----
#pragma unroll
    for (int r = 0; r < 8; ++r){
      y0[r] = (1.0f - u0[r])*n0[r] + u0[r]*y0[r];
      y1[r] = (1.0f - u1[r])*n1[r] + u1[r]*y1[r];
    }

    // ---- store latent_ys[traj, t, :] (f32, C layout; non-temporal: streamed once) ----
#pragma unroll
    for (int r = 0; r < 8; ++r){
      int m = lg*8 + r;
      size_t off = (size_t)(traj0 + m) * (TSTEPS * LATENT) + (size_t)t * LATENT;
      __builtin_nontemporal_store(y0[r], &out_ys[off + nl]);
      __builtin_nontemporal_store(y1[r], &out_ys[off + 16 + nl]);
    }

    xa = xa_n; xb = xb_n;
  }

  // ---- final state yi ----
#pragma unroll
  for (int r = 0; r < 8; ++r){
    int m = lg*8 + r;
    out_yi[(size_t)(traj0 + m) * LATENT + nl]      = y0[r];
    out_yi[(size_t)(traj0 + m) * LATENT + 16 + nl] = y1[r];
  }
}

extern "C" void kernel_launch(void* const* d_in, const int* in_sizes, int n_in,
                              void* d_out, int out_size, void* d_ws, size_t ws_size,
                              hipStream_t stream) {
  (void)in_sizes; (void)n_in; (void)out_size; (void)d_ws; (void)ws_size;
  const float* data = (const float*)d_in[0];
  // d_in[1] = time_steps (unused)
  const float* Wu1 = (const float*)d_in[2];
  const float* bu1 = (const float*)d_in[3];
  const float* Wu2 = (const float*)d_in[4];
  const float* bu2 = (const float*)d_in[5];
  const float* Wr1 = (const float*)d_in[6];
  const float* br1 = (const float*)d_in[7];
  const float* Wr2 = (const float*)d_in[8];
  const float* br2 = (const float*)d_in[9];
  const float* Wn1 = (const float*)d_in[10];
  const float* bn1 = (const float*)d_in[11];
  const float* Wn2 = (const float*)d_in[12];
  const float* bn2 = (const float*)d_in[13];

  float* out    = (float*)d_out;
  float* out_yi = out;                        // [4096, 32]
  float* out_ys = out + (size_t)NTRAJ*LATENT; // [4096, 512, 32]

  dim3 grid(NTRAJ / (TRAJ_PER_WAVE * WAVES_PER_BLOCK));  // 32 blocks
  dim3 block(WAVES_PER_BLOCK * 32);                      // 8 waves of 32
  hipLaunchKernelGGL(gru_scan_kernel, grid, block, 0, stream,
                     data, Wu1, bu1, Wu2, bu2, Wr1, br1, Wr2, br2,
                     Wn1, bn1, Wn2, bn2, out_yi, out_ys);
}
